// CoreBlock_4638564680212
// MI455X (gfx1250) — matrix operand order repaired and checked by reading, not compiled, not run on hardware
//
#include <hip/hip_runtime.h>
#include <hip/hip_bf16.h>

typedef __attribute__((ext_vector_type(16))) _Float16 v16h;
typedef __attribute__((ext_vector_type(8)))  _Float16 v8h;
typedef __attribute__((ext_vector_type(8)))  float    v8f;

#define B_  16
#define N_  1024
#define D_  768
#define H_  12
#define HS_ 64
#define M_  (B_ * N_)   // 16384 rows total

// ---------------------------------------------------------------------------
// WMMA helpers (CDNA5 v_wmma_f32_16x16x32_f16, wave32)
// ---------------------------------------------------------------------------
__device__ __forceinline__ v8f wmma16x16x32(v16h a, v16h b, v8f c) {
  return __builtin_amdgcn_wmma_f32_16x16x32_f16(false, a, false, b, (short)0, c,
                                                false, false);
}

// A fragment: 16x32 f16, row-major source, lda in elements.
// Lane L: row = m0 + (L&15); K chunks at k0 + (L>>4)*8 and +16 (ISA 7.12.2).
__device__ __forceinline__ v16h load_a_frag(const _Float16* __restrict__ A,
                                            int lda, int m0, int k0, int lane) {
  int m  = m0 + (lane & 15);
  int kb = k0 + ((lane >> 4) << 3);
  union { v16h v; v8h h[2]; } u;
  u.h[0] = *(const v8h*)(A + (size_t)m * lda + kb);
  u.h[1] = *(const v8h*)(A + (size_t)m * lda + kb + 16);
  return u.v;
}

// B fragment: 32x16 f16 from a TRANSPOSED matrix BT[col][k], ldb in elements.
// Lane L: col = c0 + (L&15); K chunks at k0 + (L>>4)*8 and +16.
__device__ __forceinline__ v16h load_b_frag(const _Float16* __restrict__ BT,
                                            int ldb, int k0, int c0, int lane) {
  int c = c0 + (lane & 15);
  int k = k0 + ((lane >> 4) << 3);
  union { v16h v; v8h h[2]; } u;
  u.h[0] = *(const v8h*)(BT + (size_t)c * ldb + k);
  u.h[1] = *(const v8h*)(BT + (size_t)c * ldb + k + 16);
  return u.v;
}

// D layout: element e of v8f -> row = e + 8*(lane>>4), col = lane&15 (+c0).

// ---------------------------------------------------------------------------
// Kernel 0: convert weights to f16, transpose to [col][k]; build alphaExt.
// ---------------------------------------------------------------------------
__global__ __launch_bounds__(256) void prep_kernel(
    const float* __restrict__ Wv, const float* __restrict__ Wf,
    const float* __restrict__ alpha, _Float16* __restrict__ WvT,
    _Float16* __restrict__ WfT, _Float16* __restrict__ aext) {
  int idx = blockIdx.x * 256 + threadIdx.x;
  if (idx < D_ * D_) {                       // WvT[c][d] = Wv[h][d][k], c=h*64+k
    int c = idx / D_, d = idx % D_;
    int h = c >> 6, k = c & 63;
    WvT[idx] = (_Float16)Wv[((size_t)h * D_ + d) * HS_ + k];
  }
  if (idx < 2 * D_ * D_) {                   // WfT[l][c][d] = Wf[l][d][c]
    int l = idx / (D_ * D_), rem = idx % (D_ * D_);
    int c = rem / D_, d = rem % D_;
    WfT[idx] = (_Float16)Wf[((size_t)l * D_ + d) * D_ + c];
  }
  if (idx < H_ * 2048) {                     // aext[h][m] = alpha[h][m & 1023]
    int h = idx >> 11, m = idx & 2047;
    aext[idx] = (_Float16)alpha[h * N_ + (m & 1023)];
  }
}

// ---------------------------------------------------------------------------
// Kernel 1: LayerNorm(x) -> xn (f16). One block per row, 256 threads.
// ---------------------------------------------------------------------------
__global__ __launch_bounds__(256) void ln_x_kernel(
    const float* __restrict__ x, const float* __restrict__ scale,
    const float* __restrict__ bias, _Float16* __restrict__ xn) {
  int row = blockIdx.x;
  const float* xr = x + (size_t)row * D_;
  int t = threadIdx.x;
  float v0 = xr[t], v1 = xr[t + 256], v2 = xr[t + 512];
  float s = v0 + v1 + v2;
  float q = v0 * v0 + v1 * v1 + v2 * v2;
  for (int off = 16; off > 0; off >>= 1) {
    s += __shfl_down(s, off, 32);
    q += __shfl_down(q, off, 32);
  }
  __shared__ float ws8[8], wq8[8];
  __shared__ float smu, srs;
  int w = t >> 5, ln = t & 31;
  if (ln == 0) { ws8[w] = s; wq8[w] = q; }
  __syncthreads();
  if (t == 0) {
    float S = 0.f, Q = 0.f;
    for (int i = 0; i < 8; ++i) { S += ws8[i]; Q += wq8[i]; }
    float mu = S * (1.0f / D_);
    float var = Q * (1.0f / D_) - mu * mu;
    smu = mu;
    srs = rsqrtf(var + 1e-6f);
  }
  __syncthreads();
  float mu = smu, rs = srs;
  _Float16* o = xn + (size_t)row * D_;
  o[t]       = (_Float16)((v0 - mu) * rs * scale[t]       + bias[t]);
  o[t + 256] = (_Float16)((v1 - mu) * rs * scale[t + 256] + bias[t + 256]);
  o[t + 512] = (_Float16)((v2 - mu) * rs * scale[t + 512] + bias[t + 512]);
}

// ---------------------------------------------------------------------------
// Kernel 2: v = xn @ WvT, output stored transposed: vT[b][c][n] (f16).
// Block = 32 rows x 768 cols; wave = 2 M-tiles x 6 N-tiles = 12 WMMA/step
// from 2 A-frag + 6 B-frag loads. K = 768 in 24 steps.
// ---------------------------------------------------------------------------
__global__ __launch_bounds__(256) void gemm_v_kernel(
    const _Float16* __restrict__ A, const _Float16* __restrict__ BT,
    _Float16* __restrict__ vT) {
  int m0   = blockIdx.x << 5;
  int wave = threadIdx.x >> 5, lane = threadIdx.x & 31;
  v8f acc[2][6];
#pragma unroll
  for (int mt = 0; mt < 2; ++mt)
#pragma unroll
    for (int t = 0; t < 6; ++t) acc[mt][t] = (v8f){0, 0, 0, 0, 0, 0, 0, 0};

  for (int k0 = 0; k0 < D_; k0 += 32) {
    v16h a0 = load_a_frag(A, D_, m0, k0, lane);
    v16h a1 = load_a_frag(A, D_, m0 + 16, k0, lane);
#pragma unroll
    for (int t = 0; t < 6; ++t) {
      int c0 = ((wave * 6 + t) << 4);
      if (k0 + 32 < D_)
        __builtin_prefetch(BT + (size_t)(c0 + (lane & 15)) * D_ + k0 + 32, 0, 1);
      v16h b = load_b_frag(BT, D_, k0, c0, lane);
      acc[0][t] = wmma16x16x32(a0, b, acc[0][t]);
      acc[1][t] = wmma16x16x32(a1, b, acc[1][t]);
    }
  }
  int bi = m0 >> 10;                          // batch (tiles never straddle)
#pragma unroll
  for (int mt = 0; mt < 2; ++mt) {
    int n0 = (m0 & 1023) + (mt << 4) + ((lane >> 4) << 3);
#pragma unroll
    for (int t = 0; t < 6; ++t) {
      int c = ((wave * 6 + t) << 4) + (lane & 15);
      v8h o;
#pragma unroll
      for (int e = 0; e < 8; ++e) o[e] = (_Float16)acc[mt][t][e];
      *(v8h*)(vT + ((size_t)bi * D_ + c) * N_ + n0) = o;
    }
  }
}

// ---------------------------------------------------------------------------
// Kernel 3: y = C @ v (circulant via alphaExt gather from LDS), fused
// residual: r = x + y (f32) and rh = f16(r).
// Block = 32 rows x 256 cols (4 heads); wave = 2 i-tiles x 2 c-tiles
// (both c-tiles in the same head) = 4 WMMA from 2 gathers + 2 B loads.
// ---------------------------------------------------------------------------
__global__ __launch_bounds__(256) void gemm_y_kernel(
    const _Float16* __restrict__ aext, const _Float16* __restrict__ vT,
    const float* __restrict__ x, float* __restrict__ r,
    _Float16* __restrict__ rh) {
  int cb = blockIdx.x << 8;                  // 256-col group
  int i0 = blockIdx.y << 5;                  // 32-row group
  int bi = blockIdx.z;
  int wave = threadIdx.x >> 5, lane = threadIdx.x & 31;

  __shared__ _Float16 lalpha[4][2048];       // the block's four heads, 16 KB
  int h0 = cb >> 6;
  for (int idx = threadIdx.x; idx < 8192; idx += 256)
    lalpha[idx >> 11][idx & 2047] = aext[((h0 + (idx >> 11)) << 11) + (idx & 2047)];
  __syncthreads();

  const _Float16* ap = lalpha[wave >> 1];    // wave w covers cols 32w..32w+31
  int c0 = cb + (wave << 5);                 // 2 c-tiles: c0, c0+16
  const _Float16* vb = vT + (size_t)bi * D_ * N_;
  v8f acc[2][2];
#pragma unroll
  for (int it = 0; it < 2; ++it)
#pragma unroll
    for (int ct = 0; ct < 2; ++ct) acc[it][ct] = (v8f){0, 0, 0, 0, 0, 0, 0, 0};

  for (int j0 = 0; j0 < N_; j0 += 32) {
    int jb = j0 + ((lane >> 4) << 3);
    v16h afr[2];
#pragma unroll
    for (int it = 0; it < 2; ++it) {
      int i    = i0 + (it << 4) + (lane & 15);
      int base = 1024 + i - jb;              // C[i][j] = aext[1024 + i - j]
      union { v16h v; _Float16 e[16]; } ua;
#pragma unroll
      for (int e = 0; e < 8; ++e) ua.e[e] = ap[base - e];
#pragma unroll
      for (int e = 0; e < 8; ++e) ua.e[8 + e] = ap[base - 16 - e];
      afr[it] = ua.v;
    }
#pragma unroll
    for (int ct = 0; ct < 2; ++ct) {
      v16h b = load_b_frag(vb, N_, j0, c0 + (ct << 4), lane);
      acc[0][ct] = wmma16x16x32(afr[0], b, acc[0][ct]);
      acc[1][ct] = wmma16x16x32(afr[1], b, acc[1][ct]);
    }
  }

#pragma unroll
  for (int it = 0; it < 2; ++it) {
    int nb = i0 + (it << 4) + ((lane >> 4) << 3);
#pragma unroll
    for (int ct = 0; ct < 2; ++ct) {
      int c = c0 + (ct << 4) + (lane & 15);
#pragma unroll
      for (int e = 0; e < 8; ++e) {
        size_t off = ((size_t)bi * N_ + nb + e) * D_ + c;
        float val = x[off] + acc[it][ct][e];
        r[off]  = val;
        rh[off] = (_Float16)val;
      }
    }
  }
}

// ---------------------------------------------------------------------------
// Kernel 4: FFN layer: t = A @ WfT + bf;  z = swish(LN(t)).
// Block = 32 rows x full 768 cols so LN stats reduce in-block (LDS).
// Wave = 2 M-tiles x 6 N-tiles. FINAL: out = log_cosh(z + r) -> d_out.
// ---------------------------------------------------------------------------
template <bool FINAL>
__global__ __launch_bounds__(256) void gemm_ffn_kernel(
    const _Float16* __restrict__ A, const _Float16* __restrict__ BT,
    const float* __restrict__ bias, const float* __restrict__ lnsc,
    const float* __restrict__ lnbi, const float* __restrict__ r,
    _Float16* __restrict__ zout, float* __restrict__ outf) {
  int m0   = blockIdx.x << 5;
  int wave = threadIdx.x >> 5, lane = threadIdx.x & 31;
  v8f acc[2][6];
#pragma unroll
  for (int mt = 0; mt < 2; ++mt)
#pragma unroll
    for (int t = 0; t < 6; ++t) acc[mt][t] = (v8f){0, 0, 0, 0, 0, 0, 0, 0};

  for (int k0 = 0; k0 < D_; k0 += 32) {
    v16h a0 = load_a_frag(A, D_, m0, k0, lane);
    v16h a1 = load_a_frag(A, D_, m0 + 16, k0, lane);
#pragma unroll
    for (int t = 0; t < 6; ++t) {
      int c0 = ((wave * 6 + t) << 4);
      if (k0 + 32 < D_)
        __builtin_prefetch(BT + (size_t)(c0 + (lane & 15)) * D_ + k0 + 32, 0, 1);
      v16h b = load_b_frag(BT, D_, k0, c0, lane);
      acc[0][t] = wmma16x16x32(a0, b, acc[0][t]);
      acc[1][t] = wmma16x16x32(a1, b, acc[1][t]);
    }
  }

  // bias add + per-lane partial moments over this lane's 6 columns
  float ps[2][8], pq[2][8];
#pragma unroll
  for (int mt = 0; mt < 2; ++mt)
#pragma unroll
    for (int e = 0; e < 8; ++e) { ps[mt][e] = 0.f; pq[mt][e] = 0.f; }
#pragma unroll
  for (int t = 0; t < 6; ++t) {
    int c = ((wave * 6 + t) << 4) + (lane & 15);
    float bc = bias[c];
#pragma unroll
    for (int mt = 0; mt < 2; ++mt)
#pragma unroll
      for (int e = 0; e < 8; ++e) {
        float v = acc[mt][t][e] + bc;
        acc[mt][t][e] = v;
        ps[mt][e] += v;
        pq[mt][e] += v * v;
      }
  }
  // butterfly across the 16 lanes of each half-wave (same row set)
#pragma unroll
  for (int off = 1; off < 16; off <<= 1)
#pragma unroll
    for (int mt = 0; mt < 2; ++mt)
#pragma unroll
      for (int e = 0; e < 8; ++e) {
        ps[mt][e] += __shfl_xor(ps[mt][e], off, 32);
        pq[mt][e] += __shfl_xor(pq[mt][e], off, 32);
      }
  __shared__ float psum[8][32], pssq[8][32];
  __shared__ float smu[32], srs[32];
  if ((lane & 15) == 0) {
    int half = lane >> 4;
#pragma unroll
    for (int mt = 0; mt < 2; ++mt)
#pragma unroll
      for (int e = 0; e < 8; ++e) {
        psum[wave][(mt << 4) + half * 8 + e] = ps[mt][e];
        pssq[wave][(mt << 4) + half * 8 + e] = pq[mt][e];
      }
  }
  __syncthreads();
  if (threadIdx.x < 32) {
    int row = threadIdx.x;
    float S = 0.f, Q = 0.f;
    for (int w = 0; w < 8; ++w) { S += psum[w][row]; Q += pssq[w][row]; }
    float mu = S * (1.0f / D_);
    float var = Q * (1.0f / D_) - mu * mu;
    smu[row] = mu;
    srs[row] = rsqrtf(var + 1e-6f);
  }
  __syncthreads();

#pragma unroll
  for (int mt = 0; mt < 2; ++mt)
#pragma unroll
    for (int t = 0; t < 6; ++t) {
      int c = ((wave * 6 + t) << 4) + (lane & 15);
      float g = lnsc[c], bb = lnbi[c];
#pragma unroll
      for (int e = 0; e < 8; ++e) {
        int row = (mt << 4) + e + ((lane >> 4) << 3);
        float yn = (acc[mt][t][e] - smu[row]) * srs[row] * g + bb;
        float sw = yn / (1.0f + __expf(-yn));             // swish
        size_t off = (size_t)(m0 + row) * D_ + c;
        if (FINAL) {
          float o = sw + r[off];
          float ax = fabsf(o);
          outf[off] = ax + log1pf(__expf(-2.0f * ax)) - 0.69314718055994531f;
        } else {
          zout[off] = (_Float16)sw;
        }
      }
    }
}

// ---------------------------------------------------------------------------
extern "C" void kernel_launch(void* const* d_in, const int* in_sizes, int n_in,
                              void* d_out, int out_size, void* d_ws,
                              size_t ws_size, hipStream_t stream) {
  const float* x     = (const float*)d_in[0];
  const float* ln1s  = (const float*)d_in[1];
  const float* ln1b  = (const float*)d_in[2];
  const float* Wv    = (const float*)d_in[3];
  const float* alpha = (const float*)d_in[4];
  const float* Wf    = (const float*)d_in[5];
  const float* bf    = (const float*)d_in[6];
  const float* lnfs  = (const float*)d_in[7];
  const float* lnfb  = (const float*)d_in[8];

  char* ws = (char*)d_ws;
  const size_t SZ16 = (size_t)M_ * D_ * 2;   // 24 MB f16 activation
  const size_t SZ32 = (size_t)M_ * D_ * 4;   // 48 MB f32 activation
  _Float16* xn   = (_Float16*)(ws);                     // [0, 24MB)
  _Float16* vT   = (_Float16*)(ws + SZ16);              // [24, 48MB)
  float*    rbuf = (float*)(ws + 2 * SZ16);             // [48, 96MB)
  _Float16* WvT  = (_Float16*)(ws + 2 * SZ16 + SZ32);   // +1.125MB
  _Float16* WfT  = WvT + (size_t)D_ * D_;               // +2.25MB
  _Float16* aext = WfT + (size_t)2 * D_ * D_;           // +48KB

  prep_kernel<<<(2 * D_ * D_) / 256, 256, 0, stream>>>(Wv, Wf, alpha, WvT, WfT,
                                                       aext);
  ln_x_kernel<<<M_, 256, 0, stream>>>(x, ln1s, ln1b, xn);
  gemm_v_kernel<<<M_ / 32, 256, 0, stream>>>(xn, WvT, vT);

  _Float16* rh = xn;  // xn dead after gemm_v
  gemm_y_kernel<<<dim3(D_ / 256, N_ / 32, B_), 256, 0, stream>>>(aext, vT, x,
                                                                 rbuf, rh);
  _Float16* z1 = vT;  // vT dead after gemm_y
  gemm_ffn_kernel<false><<<M_ / 32, 256, 0, stream>>>(
      rh, WfT, bf, lnfs, lnfb, nullptr, z1, nullptr);
  gemm_ffn_kernel<true><<<M_ / 32, 256, 0, stream>>>(
      z1, WfT + (size_t)D_ * D_, bf + D_, lnfs + D_, lnfb + D_, rbuf, nullptr,
      (float*)d_out);
}
